// Grok1MoE_62002147885123
// MI455X (gfx1250) — matrix operand order repaired and checked by reading, not compiled, not run on hardware
//
#include <hip/hip_runtime.h>
#include <hip/hip_bf16.h>
#include <math.h>

// ---------------- problem constants ----------------
constexpr int kE = 8;        // experts
constexpr int kTOPK = 2;
constexpr int kH = 2048;     // hidden
constexpr int kI = 4096;     // intermediate
constexpr int kT = 8192;     // tokens (4 * 2048)
constexpr int kRTOT = kT * kTOPK;  // 16384 routed rows total
constexpr float kCAP = 30.0f;

// ---------------- WMMA tile constants ----------------
constexpr int BM = 32;    // token rows per block (2 x 16-row A fragments)
constexpr int BN = 128;   // output cols per block (8 waves x 16 cols)
constexpr int BK = 32;    // K per WMMA step

typedef __bf16 bf16;
typedef __attribute__((ext_vector_type(16))) __bf16 v16bf;
typedef __attribute__((ext_vector_type(8)))  __bf16 v8bf;
typedef __attribute__((ext_vector_type(4)))  __bf16 v4bf;
typedef __attribute__((ext_vector_type(8)))  float  v8f;
typedef __attribute__((ext_vector_type(4)))  float  v4f;

__device__ inline float fast_silu_mul(float g, float u) {
  // silu(g) * u with v_rcp_f32 instead of IEEE divide
  const float s = __builtin_amdgcn_rcpf(1.0f + __expf(-g));
  return g * s * u;
}

// =====================================================================
// Kernel 1: router. One wave32 per token. logits = x @ w_gate,
// tanh softcap, softmax, top-2 (ties -> lower index), count per expert.
// =====================================================================
__global__ __launch_bounds__(256) void router_kernel(
    const float* __restrict__ x, const float* __restrict__ wg,
    int* __restrict__ counts, int* __restrict__ tk_idx, float* __restrict__ tk_wt) {
  const int wave = threadIdx.x >> 5;
  const int lane = threadIdx.x & 31;
  const int t = blockIdx.x * 8 + wave;

  float acc[kE];
#pragma unroll
  for (int e = 0; e < kE; ++e) acc[e] = 0.0f;

  const float* xr = x + (size_t)t * kH;
  for (int h = lane; h < kH; h += 32) {
    const float xv = xr[h];
    const float* wr = wg + (size_t)h * kE;
    v4f w0 = *(const v4f*)(wr);
    v4f w1 = *(const v4f*)(wr + 4);
#pragma unroll
    for (int e = 0; e < 4; ++e) { acc[e] += xv * w0[e]; acc[e + 4] += xv * w1[e]; }
  }
#pragma unroll
  for (int e = 0; e < kE; ++e)
    for (int off = 16; off; off >>= 1)
      acc[e] += __shfl_xor(acc[e], off, 32);

  if (lane == 0) {
    float mx = -1e30f;
#pragma unroll
    for (int e = 0; e < kE; ++e) {
      acc[e] = kCAP * tanhf(acc[e] * (1.0f / kCAP));
      if (acc[e] > mx) mx = acc[e];
    }
    float p[kE], s = 0.0f;
#pragma unroll
    for (int e = 0; e < kE; ++e) { p[e] = __expf(acc[e] - mx); s += p[e]; }
    const float inv = 1.0f / s;
#pragma unroll
    for (int e = 0; e < kE; ++e) p[e] *= inv;

    int b0 = 0;
#pragma unroll
    for (int e = 1; e < kE; ++e) if (p[e] > p[b0]) b0 = e;
    int b1 = (b0 == 0) ? 1 : 0;
#pragma unroll
    for (int e = 0; e < kE; ++e) { if (e == b0 || e == b1) continue; if (p[e] > p[b1]) b1 = e; }

    tk_idx[t * 2 + 0] = b0; tk_wt[t * 2 + 0] = p[b0];
    tk_idx[t * 2 + 1] = b1; tk_wt[t * 2 + 1] = p[b1];
    atomicAdd(&counts[b0], 1);
    atomicAdd(&counts[b1], 1);
  }
}

// =====================================================================
// Kernel 2: exclusive scan of 8 expert counts -> row bases.
// =====================================================================
__global__ void scan_kernel(const int* __restrict__ counts, int* __restrict__ bases) {
  if (threadIdx.x == 0) {
    int s = 0;
#pragma unroll
    for (int e = 0; e < kE; ++e) { bases[e] = s; s += counts[e]; }
  }
}

// =====================================================================
// Kernel 3: compact routed (token, weight) pairs into per-expert lists.
// =====================================================================
__global__ __launch_bounds__(256) void fill_kernel(
    const int* __restrict__ tk_idx, const float* __restrict__ tk_wt,
    const int* __restrict__ bases, int* __restrict__ counts2,
    int* __restrict__ rowmap, float* __restrict__ roww) {
  const int t = blockIdx.x * 256 + threadIdx.x;
  if (t >= kT) return;
#pragma unroll
  for (int j = 0; j < kTOPK; ++j) {
    const int e = tk_idx[t * 2 + j];
    const int slot = atomicAdd(&counts2[e], 1);
    const int row = bases[e] + slot;
    rowmap[row] = t;
    roww[row] = tk_wt[t * 2 + j];
  }
}

// =====================================================================
// Kernel 4: grouped GEMM  h = silu(x@w1[e]) * (x@w3[e])  (bf16 out)
// Block tile 32M x 128N; wave wv owns cols [wv*16, wv*16+16) over all 32 rows:
// two A row-fragments share each B fragment -> 4 WMMAs per 8 ds_load_b128.
// =====================================================================
__global__ __launch_bounds__(256) void gemm13_kernel(
    const float* __restrict__ x, const float* __restrict__ w1,
    const float* __restrict__ w3, const int* __restrict__ counts,
    const int* __restrict__ bases, const int* __restrict__ rowmap,
    bf16* __restrict__ hbuf) {
  const int e = blockIdx.z;
  const int cnt = counts[e];
  const int mBase = blockIdx.y * BM;
  if (mBase >= cnt) return;
  const int n0 = blockIdx.x * BN;
  const int base = bases[e];

  __shared__ bf16 Xs[BM * BK];        // [m][k], row-major
  __shared__ bf16 W1s[BN * BK];       // [n][k], transposed for B-frag reads
  __shared__ bf16 W3s[BN * BK];
  __shared__ int toks[BM];

  const int tid = threadIdx.x;
  if (tid < BM) {
    const int r = mBase + tid;
    toks[tid] = rowmap[base + ((r < cnt) ? r : (cnt - 1))];
  }
  __syncthreads();

  const int lane = tid & 31, wv = tid >> 5;
  const int hl = lane >> 4, l16 = lane & 15;     // wave32 halves per ISA layout
  const int nOff = wv * 16;                      // 0..112

  v8f acc1[2], acc3[2];
#pragma unroll
  for (int p = 0; p < 2; ++p)
#pragma unroll
    for (int i = 0; i < 8; ++i) { acc1[p][i] = 0.0f; acc3[p][i] = 0.0f; }

  const float* w1p = w1 + (size_t)e * kH * kI;
  const float* w3p = w3 + (size_t)e * kH * kI;

  const int sm = tid >> 3, sk4 = (tid & 7) * 4;  // X staging: 4 elems/thread
  const int wk = tid >> 3, wn16 = (tid & 7) * 16; // W staging: 16 elems/thread/mat

  for (int kc = 0; kc < kH; kc += BK) {
    // ---- stage X tile (gathered rows, fp32 -> bf16) ----
    {
      const float* src = x + (size_t)toks[sm] * kH + kc + sk4;
      v4f v = *(const v4f*)src;
      v4bf b; b[0] = (bf16)v[0]; b[1] = (bf16)v[1]; b[2] = (bf16)v[2]; b[3] = (bf16)v[3];
      *(v4bf*)&Xs[sm * BK + sk4] = b;
    }
    // ---- stage W1/W3 tiles transposed ([n][k]) ----
    {
      const float* s1 = w1p + (size_t)(kc + wk) * kI + n0 + wn16;
      const float* s3 = w3p + (size_t)(kc + wk) * kI + n0 + wn16;
      __builtin_prefetch(s1 + (size_t)BK * kI, 0, 1);
      __builtin_prefetch(s3 + (size_t)BK * kI, 0, 1);
#pragma unroll
      for (int q = 0; q < 4; ++q) {
        v4f a0 = *(const v4f*)(s1 + q * 4);
        v4f c0 = *(const v4f*)(s3 + q * 4);
#pragma unroll
        for (int i = 0; i < 4; ++i) {
          W1s[(wn16 + q * 4 + i) * BK + wk] = (bf16)a0[i];
          W3s[(wn16 + q * 4 + i) * BK + wk] = (bf16)c0[i];
        }
      }
    }
    __syncthreads();

    // ---- A fragments (rows 0..15 and 16..31), 16x32 bf16, K contiguous (k = 16*hl + i, as B) ----
    v16bf a[2];
#pragma unroll
    for (int p = 0; p < 2; ++p) {
      const bf16* xr = &Xs[(p * 16 + l16) * BK];
      v8bf alo = *(const v8bf*)(xr + hl * 16);
      v8bf ahi = *(const v8bf*)(xr + hl * 16 + 8);
#pragma unroll
      for (int i = 0; i < 8; ++i) { a[p][i] = alo[i]; a[p][i + 8] = ahi[i]; }
    }
    // ---- B fragments: 32x16 bf16, lane = N, K contiguous ----
    v16bf b1f, b3f;
    {
      const bf16* b1r = &W1s[(nOff + l16) * BK + hl * 16];
      const bf16* b3r = &W3s[(nOff + l16) * BK + hl * 16];
      v8bf b1lo = *(const v8bf*)b1r, b1hi = *(const v8bf*)(b1r + 8);
      v8bf b3lo = *(const v8bf*)b3r, b3hi = *(const v8bf*)(b3r + 8);
#pragma unroll
      for (int i = 0; i < 8; ++i) {
        b1f[i] = b1lo[i]; b1f[i + 8] = b1hi[i];
        b3f[i] = b3lo[i]; b3f[i + 8] = b3hi[i];
      }
    }
    acc1[0] = __builtin_amdgcn_wmma_f32_16x16x32_bf16(false, a[0], false, b1f, (short)0, acc1[0], false, false);
    acc3[0] = __builtin_amdgcn_wmma_f32_16x16x32_bf16(false, a[0], false, b3f, (short)0, acc3[0], false, false);
    acc1[1] = __builtin_amdgcn_wmma_f32_16x16x32_bf16(false, a[1], false, b1f, (short)0, acc1[1], false, false);
    acc3[1] = __builtin_amdgcn_wmma_f32_16x16x32_bf16(false, a[1], false, b3f, (short)0, acc3[1], false, false);
    __syncthreads();
  }

  // ---- epilogue: h = silu(g) * u, store bf16 ----
#pragma unroll
  for (int p = 0; p < 2; ++p) {
#pragma unroll
    for (int r = 0; r < 8; ++r) {
      const int m = p * 16 + hl * 8 + r;
      const float hval = fast_silu_mul(acc1[p][r], acc3[p][r]);
      if (mBase + m < cnt) {
        const size_t row = (size_t)(base + mBase + m);
        hbuf[row * kI + n0 + nOff + l16] = (bf16)hval;
      }
    }
  }
}

// =====================================================================
// Kernel 5: grouped GEMM  out[token] += (h @ w2[e]) * routing_weight
// A fragments loaded directly from bf16 hbuf (already in WMMA layout);
// B fragment reused across both row halves -> 1 ds_load per WMMA.
// =====================================================================
__global__ __launch_bounds__(256) void gemm2_kernel(
    const bf16* __restrict__ hbuf, const float* __restrict__ w2,
    const int* __restrict__ counts, const int* __restrict__ bases,
    const int* __restrict__ rowmap, const float* __restrict__ roww,
    float* __restrict__ out) {
  const int e = blockIdx.z;
  const int cnt = counts[e];
  const int mBase = blockIdx.y * BM;
  if (mBase >= cnt) return;
  const int n0 = blockIdx.x * BN;  // over H
  const int base = bases[e];

  __shared__ bf16 W2s[BN * BK];    // [n][k] transposed

  const int tid = threadIdx.x, lane = tid & 31, wv = tid >> 5;
  const int hl = lane >> 4, l16 = lane & 15;
  const int nOff = wv * 16;

  v8f acc[2];
#pragma unroll
  for (int p = 0; p < 2; ++p)
#pragma unroll
    for (int i = 0; i < 8; ++i) acc[p][i] = 0.0f;

  const float* w2p = w2 + (size_t)e * kI * kH;

  const bf16* hrow[2];
#pragma unroll
  for (int p = 0; p < 2; ++p) {
    int rl = mBase + p * 16 + l16;
    if (rl >= cnt) rl = cnt - 1;              // clamp: garbage rows never stored
    hrow[p] = hbuf + (size_t)(base + rl) * kI;
  }

  const int wk = tid >> 3, wn16 = (tid & 7) * 16;

  for (int kc = 0; kc < kI; kc += BK) {
    // ---- stage W2 tile transposed ([n][k]) ----
    {
      const float* s = w2p + (size_t)(kc + wk) * kH + n0 + wn16;
      __builtin_prefetch(s + (size_t)BK * kH, 0, 1);
#pragma unroll
      for (int q = 0; q < 4; ++q) {
        v4f a0 = *(const v4f*)(s + q * 4);
#pragma unroll
        for (int i = 0; i < 4; ++i)
          W2s[(wn16 + q * 4 + i) * BK + wk] = (bf16)a0[i];
      }
    }
    __syncthreads();

    v16bf b;
    {
      const bf16* br = &W2s[(nOff + l16) * BK + hl * 16];
      v8bf blo = *(const v8bf*)br, bhi = *(const v8bf*)(br + 8);
#pragma unroll
      for (int i = 0; i < 8; ++i) { b[i] = blo[i]; b[i + 8] = bhi[i]; }
    }
#pragma unroll
    for (int p = 0; p < 2; ++p) {
      v16bf a;
      const bf16* ar = hrow[p] + kc;
      v8bf alo = *(const v8bf*)(ar + hl * 16);
      v8bf ahi = *(const v8bf*)(ar + hl * 16 + 8);
#pragma unroll
      for (int i = 0; i < 8; ++i) { a[i] = alo[i]; a[i + 8] = ahi[i]; }
      acc[p] = __builtin_amdgcn_wmma_f32_16x16x32_bf16(false, a, false, b, (short)0, acc[p], false, false);
    }
    __syncthreads();
  }

#pragma unroll
  for (int p = 0; p < 2; ++p) {
#pragma unroll
    for (int r = 0; r < 8; ++r) {
      const int m = p * 16 + hl * 8 + r;
      const int rl = mBase + m;
      if (rl < cnt) {
        const float w = roww[base + rl];
        const int t = rowmap[base + rl];
        atomicAdd(&out[(size_t)t * kH + n0 + nOff + l16], acc[p][r] * w);
      }
    }
  }
}

// =====================================================================
// host launcher
// =====================================================================
extern "C" void kernel_launch(void* const* d_in, const int* in_sizes, int n_in,
                              void* d_out, int out_size, void* d_ws, size_t ws_size,
                              hipStream_t stream) {
  const float* x  = (const float*)d_in[0];   // [T, H]
  const float* wg = (const float*)d_in[1];   // [H, E]
  const float* w1 = (const float*)d_in[2];   // [E, H, I]
  const float* w3 = (const float*)d_in[3];   // [E, H, I]
  const float* w2 = (const float*)d_in[4];   // [E, I, H]
  float* out = (float*)d_out;                // [T, H]

  char* ws = (char*)d_ws;
  int*   counts  = (int*)(ws + 0);
  int*   counts2 = (int*)(ws + 256);
  int*   bases   = (int*)(ws + 512);
  int*   tk_idx  = (int*)(ws + 1024);
  float* tk_wt   = (float*)(ws + 1024 + (size_t)kRTOT * 4);
  int*   rowmap  = (int*)(ws + 1024 + (size_t)kRTOT * 8);
  float* roww    = (float*)(ws + 1024 + (size_t)kRTOT * 12);
  bf16*  hbuf    = (bf16*)(ws + (1 << 20));  // 16384 x 4096 bf16 = 128 MB

  (void)in_sizes; (void)n_in; (void)ws_size;

  hipMemsetAsync(ws, 0, 1024, stream);                                   // counts, counts2, bases
  hipMemsetAsync(d_out, 0, (size_t)out_size * sizeof(float), stream);    // fp32 accumulation target

  router_kernel<<<kT / 8, 256, 0, stream>>>(x, wg, counts, tk_idx, tk_wt);
  scan_kernel<<<1, 32, 0, stream>>>(counts, bases);
  fill_kernel<<<kT / 256, 256, 0, stream>>>(tk_idx, tk_wt, bases, counts2, rowmap, roww);

  gemm13_kernel<<<dim3(kI / BN, 256, kE), 256, 0, stream>>>(
      x, w1, w3, counts, bases, rowmap, hbuf);
  gemm2_kernel<<<dim3(kH / BN, 256, kE), 256, 0, stream>>>(
      hbuf, w2, counts, bases, rowmap, roww, out);
}
